// IterativeGNN_89008902242871
// MI455X (gfx1250) — matrix operand order repaired and checked by reading, not compiled, not run on hardware
//
#include <hip/hip_runtime.h>

#define NN 20000
#define EE 320000
#define HH 128

typedef __attribute__((ext_vector_type(16))) _Float16 v16h;
typedef __attribute__((ext_vector_type(8)))  float    v8f;

// ---------------------------------------------------------------- WMMA helpers

// A-fragment (16x32 f16, M x K) from LDS row-major [rows x lda], row offset m0.
// Per ISA 7.12.2: lanes 0-15 hold row M=lane with K = kbase+{0..7, 16..23},
// lanes 16-31 hold row M=lane-16 with K = kbase+{8..15, 24..31}.
__device__ __forceinline__ v16h wmma_load_a_lds(const _Float16* smem, int lda,
                                                int m0, int kbase) {
  const int l = threadIdx.x & 31;
  const int m = l & 15;
  const int half = l >> 4;
  const _Float16* p = smem + (m0 + m) * lda + kbase + half * 8;
  v16h a;
#pragma unroll
  for (int i = 0; i < 8; ++i) { a[i] = p[i]; a[8 + i] = p[16 + i]; }
  return a;
}

// B-fragment from pre-swizzled weights: each (colblk*KT + kt) fragment is
// stored as 32 lanes x 16 contiguous f16 -> one coalesced 32B load per lane.
__device__ __forceinline__ v16h wmma_load_b_sw(const _Float16* Wsw, int frag) {
  const int l = threadIdx.x & 31;
  return *(const v16h*)(Wsw + ((size_t)(frag * 32 + l) << 4));
}

__device__ __forceinline__ v8f wmma_step(v16h a, v16h b, v8f c) {
  return __builtin_amdgcn_wmma_f32_16x16x32_f16(false, a, false, b, (short)0, c, false, false);
}

// ------------------------------------------------------- weight pre-swizzling
// Dest layout: dst = ((c*KT + kt)*32 + lane)*16 + i
//   row = kt*32 + (lane>>4)*8 + (i&7) + (i>>3)*16   (B fragment: lanes 0-15 K {0..7,16..23}, 16-31 K {8..15,24..31})
//   col = c*16 + (lane&15)
// Rows >= 128 come from Wb (stacked SAGE Wl|Wr, K=256).
__global__ void k_swz(const float* __restrict__ Wa, const float* __restrict__ Wb,
                      _Float16* __restrict__ o, int KT) {
  const int total = KT * 32 * 16 * 8;  // K * 128
  for (int d = blockIdx.x * blockDim.x + threadIdx.x; d < total;
       d += gridDim.x * blockDim.x) {
    int i   = d & 15;
    int l   = (d >> 4) & 31;
    int ktc = d >> 9;
    int kt  = ktc % KT;
    int c   = ktc / KT;
    int row = kt * 32 + (l >> 4) * 8 + (i & 7) + (i >> 3) * 16;
    int col = c * 16 + (l & 15);
    float v = (row < 128) ? Wa[row * 128 + col] : Wb[(row - 128) * 128 + col];
    o[d] = (_Float16)v;
  }
}

// ------------------------------------------------------------------ utilities
__global__ void k_zero(float* p, int n) {
  int i = blockIdx.x * blockDim.x + threadIdx.x;
  if (i < n) p[i] = 0.0f;
}

__global__ void k_node_features(const float* __restrict__ shapes,
                                const int* __restrict__ gate_ids,
                                const float* __restrict__ emb,
                                const float* __restrict__ dimW,
                                const float* __restrict__ dimb,
                                float* __restrict__ x) {
  int idx = blockIdx.x * blockDim.x + threadIdx.x;
  if (idx >= NN * HH) return;
  int n = idx >> 7, h = idx & 127;
  float v;
  if (h < 64) v = shapes[n] * dimW[h] + dimb[h];
  else        v = emb[gate_ids[n] * 64 + (h - 64)];
  x[idx] = v;
}

// ---------------------------------------------------------------- SAGE layers
__global__ void k_sage_aggr(const float* __restrict__ x, const int* __restrict__ src,
                            const int* __restrict__ dst, float* __restrict__ aggr,
                            float* __restrict__ cnt) {
  int e = blockIdx.x * (blockDim.x >> 5) + (threadIdx.x >> 5);
  if (e >= EE) return;
  int lane = threadIdx.x & 31;
  const float* xs = x + (size_t)src[e] * HH;
  float* ad = aggr + (size_t)dst[e] * HH;
#pragma unroll
  for (int c = lane; c < HH; c += 32) atomicAdd(&ad[c], xs[c]);
  if (lane == 0) atomicAdd(&cnt[dst[e]], 1.0f);
}

// x_new = relu([mean_aggr | x] @ [Wl;Wr] + bl)
// 32 nodes/WG, K=256 stacked, 8 waves; each wave: 2 M-subtiles sharing B frags.
__global__ void __launch_bounds__(256)
k_sage_gemm(const float* __restrict__ x, const float* __restrict__ aggr,
            const float* __restrict__ cnt, const _Float16* __restrict__ Wsw,
            const float* __restrict__ bias, float* __restrict__ xout) {
  __shared__ __align__(16) _Float16 sA[32 * 256];
  const int node0 = blockIdx.x * 32;
  for (int idx = threadIdx.x; idx < 32 * 256; idx += 256) {
    int r = idx >> 8, k = idx & 255;
    int n = node0 + r;
    float v;
    if (k < 128) v = aggr[(size_t)n * HH + k] / fmaxf(cnt[n], 1.0f);
    else         v = x[(size_t)n * HH + (k - 128)];
    sA[idx] = (_Float16)v;
  }
  __syncthreads();
  const int w = threadIdx.x >> 5;   // column block 0..7
  v8f acc0 = {}, acc1 = {};
#pragma unroll
  for (int kt = 0; kt < 8; ++kt) {
    v16h b  = wmma_load_b_sw(Wsw, w * 8 + kt);
    v16h a0 = wmma_load_a_lds(sA, 256, 0,  kt * 32);
    v16h a1 = wmma_load_a_lds(sA, 256, 16, kt * 32);
    acc0 = wmma_step(a0, b, acc0);   // back-to-back WMMA, shared B operand
    acc1 = wmma_step(a1, b, acc1);
  }
  const int l = threadIdx.x & 31;
  const int col = w * 16 + (l & 15);
  const int mhi = (l >> 4) * 8;
  const float bv = bias[col];
#pragma unroll
  for (int v = 0; v < 8; ++v) {
    xout[(size_t)(node0 + mhi + v) * HH + col]      = fmaxf(acc0[v] + bv, 0.0f);
    xout[(size_t)(node0 + 16 + mhi + v) * HH + col] = fmaxf(acc1[v] + bv, 0.0f);
  }
}

// --------------------------------------------------------- generic node GEMM
// 32 nodes/WG, K=128; used for the GAT projection h = x @ W.
__global__ void __launch_bounds__(256)
k_node_gemm(const float* __restrict__ x, const _Float16* __restrict__ Wsw,
            const float* __restrict__ bias, float* __restrict__ out, int do_relu) {
  __shared__ __align__(16) _Float16 sA[32 * 128];
  const int node0 = blockIdx.x * 32;
  for (int idx = threadIdx.x; idx < 32 * 128; idx += 256) {
    int r = idx >> 7, k = idx & 127;
    sA[idx] = (_Float16)x[(size_t)(node0 + r) * HH + k];
  }
  __syncthreads();
  const int w = threadIdx.x >> 5;
  v8f acc0 = {}, acc1 = {};
#pragma unroll
  for (int kt = 0; kt < 4; ++kt) {
    v16h b  = wmma_load_b_sw(Wsw, w * 4 + kt);
    v16h a0 = wmma_load_a_lds(sA, 128, 0,  kt * 32);
    v16h a1 = wmma_load_a_lds(sA, 128, 16, kt * 32);
    acc0 = wmma_step(a0, b, acc0);
    acc1 = wmma_step(a1, b, acc1);
  }
  const int l = threadIdx.x & 31;
  const int col = w * 16 + (l & 15);
  const int mhi = (l >> 4) * 8;
  const float bv = bias ? bias[col] : 0.0f;
#pragma unroll
  for (int v = 0; v < 8; ++v) {
    float r0 = acc0[v] + bv, r1 = acc1[v] + bv;
    if (do_relu) { r0 = fmaxf(r0, 0.0f); r1 = fmaxf(r1, 0.0f); }
    out[(size_t)(node0 + mhi + v) * HH + col]      = r0;
    out[(size_t)(node0 + 16 + mhi + v) * HH + col] = r1;
  }
}

// ------------------------------------------------------------------ GAT layer
__device__ __forceinline__ unsigned fenc(float f) {
  unsigned u = __float_as_uint(f);
  return (u & 0x80000000u) ? ~u : (u | 0x80000000u);
}
__device__ __forceinline__ float fdec(unsigned u) {
  return (u & 0x80000000u) ? __uint_as_float(u & 0x7fffffffu) : __uint_as_float(~u);
}

__global__ void k_gat_nodescal(const float* __restrict__ h, const float* __restrict__ asrc,
                               const float* __restrict__ adst, float* __restrict__ as_,
                               float* __restrict__ ad_) {
  int n = blockIdx.x * (blockDim.x >> 5) + (threadIdx.x >> 5);
  if (n >= NN) return;
  int lane = threadIdx.x & 31;
  const float* hr = h + (size_t)n * HH;
  float s0 = 0.f, s1 = 0.f;
#pragma unroll
  for (int c = lane; c < HH; c += 32) { float v = hr[c]; s0 += v * asrc[c]; s1 += v * adst[c]; }
#pragma unroll
  for (int off = 16; off > 0; off >>= 1) { s0 += __shfl_xor(s0, off, 32); s1 += __shfl_xor(s1, off, 32); }
  if (lane == 0) { as_[n] = s0; ad_[n] = s1; }
}

__global__ void k_gat_e(const int* __restrict__ src, const int* __restrict__ dst,
                        const float* __restrict__ as_, const float* __restrict__ ad_,
                        float* __restrict__ earr, unsigned* __restrict__ menc) {
  int i = blockIdx.x * blockDim.x + threadIdx.x;
  if (i >= EE + NN) return;
  int s = (i < EE) ? src[i] : (i - EE);
  int d = (i < EE) ? dst[i] : (i - EE);
  float v = as_[s] + ad_[d];
  float e = (v > 0.f) ? v : 0.2f * v;   // leaky_relu(0.2)
  earr[i] = e;
  atomicMax(&menc[d], fenc(e));         // encoded fp-max; 0-init is below all reals
}

__global__ void k_gat_p(const int* __restrict__ dst, const float* __restrict__ earr,
                        const unsigned* __restrict__ menc, float* __restrict__ parr,
                        float* __restrict__ z) {
  int i = blockIdx.x * blockDim.x + threadIdx.x;
  if (i >= EE + NN) return;
  int d = (i < EE) ? dst[i] : (i - EE);
  float p = __expf(earr[i] - fdec(menc[d]));
  parr[i] = p;
  atomicAdd(&z[d], p);
}

__global__ void k_gat_scatter(const int* __restrict__ src, const int* __restrict__ dst,
                              const float* __restrict__ h, const float* __restrict__ parr,
                              const float* __restrict__ z, float* __restrict__ acc) {
  int i = blockIdx.x * (blockDim.x >> 5) + (threadIdx.x >> 5);
  if (i >= EE + NN) return;
  int lane = threadIdx.x & 31;
  int s = (i < EE) ? src[i] : (i - EE);
  int d = (i < EE) ? dst[i] : (i - EE);
  float alpha = parr[i] / z[d];
  const float* hs = h + (size_t)s * HH;
  float* ad = acc + (size_t)d * HH;
#pragma unroll
  for (int c = lane; c < HH; c += 32) atomicAdd(&ad[c], alpha * hs[c]);
}

__global__ void k_gat_finish(const float* __restrict__ acc, const float* __restrict__ b,
                             float* __restrict__ x) {
  int idx = blockIdx.x * blockDim.x + threadIdx.x;
  if (idx >= NN * HH) return;
  x[idx] = fmaxf(acc[idx] + b[idx & 127], 0.0f);
}

// ------------------------------------------- fused edge MLP (dominant GEMMs)
// 32 edges/WG; both MLP layers + score fused, no global intermediates.
__global__ void __launch_bounds__(256)
k_edge_mlp(const float* __restrict__ x, const int* __restrict__ src,
           const int* __restrict__ dst, const _Float16* __restrict__ W0sw,
           const float* __restrict__ b0, const _Float16* __restrict__ W1sw,
           const float* __restrict__ b1, const float* __restrict__ outW,
           const float* __restrict__ outb, float* __restrict__ scores) {
  __shared__ __align__(16) _Float16 sA[32 * 128];
  __shared__ __align__(16) _Float16 sB[32 * 128];
  const int e0 = blockIdx.x * 32;
  for (int idx = threadIdx.x; idx < 32 * 128; idx += 256) {
    int r = idx >> 7, k = idx & 127;
    int e = e0 + r;
    sA[idx] = (_Float16)(x[(size_t)src[e] * HH + k] + x[(size_t)dst[e] * HH + k]);
  }
  __syncthreads();
  const int w = threadIdx.x >> 5;
  const int l = threadIdx.x & 31;
  const int col = w * 16 + (l & 15);
  const int mhi = (l >> 4) * 8;
  // layer 0: sA -> sB
  {
    v8f acc0 = {}, acc1 = {};
#pragma unroll
    for (int kt = 0; kt < 4; ++kt) {
      v16h b  = wmma_load_b_sw(W0sw, w * 4 + kt);
      v16h a0 = wmma_load_a_lds(sA, 128, 0,  kt * 32);
      v16h a1 = wmma_load_a_lds(sA, 128, 16, kt * 32);
      acc0 = wmma_step(a0, b, acc0);
      acc1 = wmma_step(a1, b, acc1);
    }
    float bv = b0[col];
#pragma unroll
    for (int v = 0; v < 8; ++v) {
      sB[(mhi + v) * 128 + col]      = (_Float16)fmaxf(acc0[v] + bv, 0.0f);
      sB[(16 + mhi + v) * 128 + col] = (_Float16)fmaxf(acc1[v] + bv, 0.0f);
    }
  }
  __syncthreads();
  // layer 1: sB -> sA
  {
    v8f acc0 = {}, acc1 = {};
#pragma unroll
    for (int kt = 0; kt < 4; ++kt) {
      v16h b  = wmma_load_b_sw(W1sw, w * 4 + kt);
      v16h a0 = wmma_load_a_lds(sB, 128, 0,  kt * 32);
      v16h a1 = wmma_load_a_lds(sB, 128, 16, kt * 32);
      acc0 = wmma_step(a0, b, acc0);
      acc1 = wmma_step(a1, b, acc1);
    }
    float bv = b1[col];
#pragma unroll
    for (int v = 0; v < 8; ++v) {
      sA[(mhi + v) * 128 + col]      = (_Float16)fmaxf(acc0[v] + bv, 0.0f);
      sA[(16 + mhi + v) * 128 + col] = (_Float16)fmaxf(acc1[v] + bv, 0.0f);
    }
  }
  __syncthreads();
  // final score: 128-length dot with outW (f32); one row per thread 0..31
  if (threadIdx.x < 32) {
    const _Float16* row = sA + threadIdx.x * 128;
    float s = 0.0f;
#pragma unroll 8
    for (int i = 0; i < 128; ++i) s += (float)row[i] * outW[i];
    scores[e0 + threadIdx.x] = s + outb[0];
  }
}

// -------------------------------------------------------------------- launch
static inline int cdiv(int a, int b) { return (a + b - 1) / b; }

extern "C" void kernel_launch(void* const* d_in, const int* in_sizes, int n_in,
                              void* d_out, int out_size, void* d_ws, size_t ws_size,
                              hipStream_t stream) {
  const float* shapes   = (const float*)d_in[0];
  const int*   gate_ids = (const int*)d_in[1];
  const int*   eidx     = (const int*)d_in[2];
  const float* emb      = (const float*)d_in[3];
  const float* dimW     = (const float*)d_in[4];
  const float* dimb     = (const float*)d_in[5];
  const float* sage_Wl  = (const float*)d_in[6];
  const float* sage_bl  = (const float*)d_in[7];
  const float* sage_Wr  = (const float*)d_in[8];
  const float* gat_W    = (const float*)d_in[9];
  const float* gat_asrc = (const float*)d_in[10];
  const float* gat_adst = (const float*)d_in[11];
  const float* gat_b    = (const float*)d_in[12];
  const float* edgeW    = (const float*)d_in[13];
  const float* edgeb    = (const float*)d_in[14];
  const float* outW     = (const float*)d_in[15];
  const float* outb     = (const float*)d_in[16];
  const int* src = eidx;
  const int* dst = eidx + EE;

  char* p = (char*)d_ws;
  auto alloc = [&](size_t bytes) -> void* {
    void* r = (void*)p;
    p += (bytes + 255) & ~(size_t)255;
    return r;
  };
  float*     xA     = (float*)alloc((size_t)NN * HH * 4);
  float*     xB     = (float*)alloc((size_t)NN * HH * 4);
  float*     aggr   = (float*)alloc((size_t)NN * HH * 4);   // reused as GAT h
  float*     gacc   = (float*)alloc((size_t)NN * HH * 4);
  float*     cnt    = (float*)alloc((size_t)NN * 4);
  float*     as_    = (float*)alloc((size_t)NN * 4);
  float*     ad_    = (float*)alloc((size_t)NN * 4);
  unsigned*  menc   = (unsigned*)alloc((size_t)NN * 4);
  float*     zbuf   = (float*)alloc((size_t)NN * 4);
  float*     earr   = (float*)alloc((size_t)(EE + NN) * 4);
  float*     parr   = (float*)alloc((size_t)(EE + NN) * 4);
  _Float16*  sageSw = (_Float16*)alloc((size_t)3 * 256 * 128 * 2);
  _Float16*  gatSw  = (_Float16*)alloc((size_t)128 * 128 * 2);
  _Float16*  w0Sw   = (_Float16*)alloc((size_t)128 * 128 * 2);
  _Float16*  w1Sw   = (_Float16*)alloc((size_t)128 * 128 * 2);

  const int T = 256;
  // pre-swizzle all weights into WMMA B-fragment layout (f16)
  for (int i = 0; i < 3; ++i)
    k_swz<<<cdiv(256 * 128, T), T, 0, stream>>>(sage_Wl + i * 128 * 128,
                                                sage_Wr + i * 128 * 128,
                                                sageSw + i * 256 * 128, 8);
  k_swz<<<cdiv(128 * 128, T), T, 0, stream>>>(gat_W, gat_W, gatSw, 4);
  k_swz<<<cdiv(128 * 128, T), T, 0, stream>>>(edgeW, edgeW, w0Sw, 4);
  k_swz<<<cdiv(128 * 128, T), T, 0, stream>>>(edgeW + 128 * 128, edgeW + 128 * 128, w1Sw, 4);

  // initial node features
  k_node_features<<<cdiv(NN * HH, T), T, 0, stream>>>(shapes, gate_ids, emb, dimW, dimb, xA);

  float* xcur = xA;
  float* xnext = xB;
  for (int li = 0; li < 3; ++li) {
    k_zero<<<cdiv(NN * HH, T), T, 0, stream>>>(aggr, NN * HH);
    k_zero<<<cdiv(NN, T), T, 0, stream>>>(cnt, NN);
    k_sage_aggr<<<cdiv(EE, 8), T, 0, stream>>>(xcur, src, dst, aggr, cnt);
    k_sage_gemm<<<NN / 32, T, 0, stream>>>(xcur, aggr, cnt, sageSw + li * 256 * 128,
                                           sage_bl + li * 128, xnext);
    float* t = xcur; xcur = xnext; xnext = t;
  }

  // GAT layer
  float* h = aggr;
  k_node_gemm<<<NN / 32, T, 0, stream>>>(xcur, gatSw, nullptr, h, 0);
  k_gat_nodescal<<<cdiv(NN, 8), T, 0, stream>>>(h, gat_asrc, gat_adst, as_, ad_);
  k_zero<<<cdiv(NN, T), T, 0, stream>>>((float*)menc, NN);
  k_zero<<<cdiv(NN, T), T, 0, stream>>>(zbuf, NN);
  k_zero<<<cdiv(NN * HH, T), T, 0, stream>>>(gacc, NN * HH);
  k_gat_e<<<cdiv(EE + NN, T), T, 0, stream>>>(src, dst, as_, ad_, earr, menc);
  k_gat_p<<<cdiv(EE + NN, T), T, 0, stream>>>(dst, earr, menc, parr, zbuf);
  k_gat_scatter<<<cdiv(EE + NN, 8), T, 0, stream>>>(src, dst, h, parr, zbuf, gacc);
  k_gat_finish<<<cdiv(NN * HH, T), T, 0, stream>>>(gacc, gat_b, xnext);
  { float* t = xcur; xcur = xnext; xnext = t; }

  // fused edge MLP -> scores
  k_edge_mlp<<<EE / 32, T, 0, stream>>>(xcur, src, dst, w0Sw, edgeb, w1Sw, edgeb + 128,
                                        outW, outb, (float*)d_out);
}
